// MambaFormerBackbone_77421080478523
// MI455X (gfx1250) — matrix operand order repaired and checked by reading, not compiled, not run on hardware
//
#include <hip/hip_runtime.h>

// ---------------------------------------------------------------------------
// MambaFormer backbone for gfx1250 (MI455X). GEMMs use v_wmma_f32_16x16x32_bf16
// (bf16 inputs, f32 accumulate). Scan / norms / conv / attention in f32 VALU.
// ---------------------------------------------------------------------------

#define DMODEL   768
#define BATCH    2
#define SEQ      512
#define LCOND    64
#define ROWS     (BATCH * SEQ)          // 1024 tokens
#define CROWS    (BATCH * LCOND)        // 128 cond rows
#define DINNER   1536
#define DSTATE   128
#define HEADDIM  64
#define NHEADS   24
#define CONVDIM  1792                   // DINNER + 2*DSTATE
#define DINPROJ  3352                   // 2*DINNER + 2*DSTATE + NHEADS
#define AHEADS   8
#define ADIM     64
#define AINNER   512
#define NLAYER   8
#define AFFD     (6 * DMODEL)           // 4608
#define MT       4                      // M-tiles (16 rows each) per wave

typedef __attribute__((ext_vector_type(16))) __bf16 v16bf;
typedef __attribute__((ext_vector_type(8)))  float  v8f;

__device__ __forceinline__ float sigm(float x) { return 1.0f / (1.0f + __expf(-x)); }
__device__ __forceinline__ __bf16 f2bf(float x) { return (__bf16)x; }

// ---------------------------------------------------------------------------
// WMMA GEMM: C[M,N] = A[M,K] * W[K,N] (+bias). A,W row-major f32, converted to
// bf16 fragments. One wave computes a 64x16 tile (4 WMMA accumulators sharing
// one B fragment per K-step). K % 32 == 0; M % 64 == 0; N arbitrary (branch-
// free column clamp + select).
// Fragment layouts used here (wave32; A and B share contiguous k order):
//   A (16x32): lane l -> row M=l%16 ; slot e -> K = e + 16*(l/16)
//              => per lane one contiguous 16-float span: [kk+16g, +16)
//   B (32x16): lane l -> col N=l%16 ; slot e -> K = e + 16*(l/16)
//   C/D      : lane l -> col N=l%16 ; vgpr r -> row M = r + 8*(l/16)
// ---------------------------------------------------------------------------
__global__ void gemm_bf16_wmma_k(const float* __restrict__ A,
                                 const float* __restrict__ W,
                                 const float* __restrict__ bias,
                                 float* __restrict__ C,
                                 int M, int N, int K) {
  const int n0   = blockIdx.x * 16;
  const int m0   = blockIdx.y * (16 * MT);
  const int lane = threadIdx.x;       // 0..31
  const int g    = lane >> 4;         // half-wave group
  const int mi   = lane & 15;
  const int col  = n0 + mi;
  const bool cok = (col < N);
  const int colc = cok ? col : (N - 1);   // clamped: load always in bounds

  v8f acc[MT] = {};

  for (int kk = 0; kk < K; kk += 32) {
    // ---- B fragment: 16 column-strided loads, unconditional, value-selected
    v16bf b;
    const float* wp = W + (size_t)(kk + 16 * g) * N + colc;
#pragma unroll
    for (int e = 0; e < 16; ++e) {
      float w = wp[(size_t)e * N];
      b[e] = f2bf(cok ? w : 0.0f);
    }
#pragma unroll
    for (int t = 0; t < MT; ++t) {
      // ---- A fragment: 4x float4 from one contiguous 16-float span
      const float* ap = A + (size_t)(m0 + t * 16 + mi) * K + kk + 16 * g;
      float4 a0 = *(const float4*)(ap);
      float4 a1 = *(const float4*)(ap + 4);
      float4 a2 = *(const float4*)(ap + 8);
      float4 a3 = *(const float4*)(ap + 12);
      v16bf a;
      a[0]  = f2bf(a0.x); a[1]  = f2bf(a0.y); a[2]  = f2bf(a0.z); a[3]  = f2bf(a0.w);
      a[4]  = f2bf(a1.x); a[5]  = f2bf(a1.y); a[6]  = f2bf(a1.z); a[7]  = f2bf(a1.w);
      a[8]  = f2bf(a2.x); a[9]  = f2bf(a2.y); a[10] = f2bf(a2.z); a[11] = f2bf(a2.w);
      a[12] = f2bf(a3.x); a[13] = f2bf(a3.y); a[14] = f2bf(a3.z); a[15] = f2bf(a3.w);
      acc[t] = __builtin_amdgcn_wmma_f32_16x16x32_bf16(
          false, a, false, b, (short)0, acc[t], false, false);
    }
  }

  if (cok) {
    const float bs = bias ? bias[col] : 0.0f;
#pragma unroll
    for (int t = 0; t < MT; ++t) {
#pragma unroll
      for (int r = 0; r < 8; ++r) {
        const int row = m0 + t * 16 + r + 8 * g;
        C[(size_t)row * N + col] = acc[t][r] + bs;
      }
    }
  }
}

// --------------------------- pooled cond (once) ----------------------------
__global__ void pool_cond_k(const float* __restrict__ cond, float* __restrict__ pooled) {
  int idx = blockIdx.x * blockDim.x + threadIdx.x;
  if (idx >= BATCH * DMODEL) return;
  int b = idx / DMODEL, d = idx % DMODEL;
  float s = 0.0f;
  for (int l = 0; l < LCOND; ++l) s += cond[(b * LCOND + l) * DMODEL + d];
  s *= (1.0f / (float)LCOND);
  pooled[idx] = s * sigm(s);
}

// --------------------------- aff = pooled @ ada_w + ada_b ------------------
__global__ void aff_k(const float* __restrict__ pooled,
                      const float* __restrict__ ada_w,
                      const float* __restrict__ ada_b,
                      float* __restrict__ aff) {
  int idx = blockIdx.x * blockDim.x + threadIdx.x;
  if (idx >= BATCH * AFFD) return;
  int b = idx / AFFD, j = idx % AFFD;
  float s = ada_b[j];
  for (int d = 0; d < DMODEL; ++d) s += pooled[b * DMODEL + d] * ada_w[(size_t)d * AFFD + j];
  aff[idx] = s;
}

// ---- residual = hidden+residual; hidden = rmsnorm(residual)*w; u = mod ----
__global__ void addnorm_mod_k(float* __restrict__ hidden,
                              float* __restrict__ residual,
                              const float* __restrict__ norm_w,
                              const float* __restrict__ aff,
                              float* __restrict__ u_mba) {
  __shared__ float red[256];
  const int row = blockIdx.x;
  const int b   = row >> 9;
  const float* sh = aff + b * AFFD;           // sh_mba
  const float* sc = aff + b * AFFD + DMODEL;  // sc_mba
  float v[3]; float ss = 0.0f;
#pragma unroll
  for (int i = 0; i < 3; ++i) {
    int d = threadIdx.x + i * 256;
    float x = hidden[row * DMODEL + d] + residual[row * DMODEL + d];
    residual[row * DMODEL + d] = x;
    v[i] = x; ss += x * x;
  }
  red[threadIdx.x] = ss; __syncthreads();
  for (int s = 128; s > 0; s >>= 1) {
    if (threadIdx.x < s) red[threadIdx.x] += red[threadIdx.x + s];
    __syncthreads();
  }
  float inv = rsqrtf(red[0] / (float)DMODEL + 1e-5f);
#pragma unroll
  for (int i = 0; i < 3; ++i) {
    int d = threadIdx.x + i * 256;
    float hn = v[i] * inv * norm_w[d];
    hidden[row * DMODEL + d] = hn;
    u_mba[row * DMODEL + d]  = hn * (1.0f + sc[d]) + sh[d];
  }
}

// ------------------------- dt = softplus, dA = exp -------------------------
__global__ void dt_k(const float* __restrict__ zx,
                     const float* __restrict__ dt_bias,
                     const float* __restrict__ A_log,
                     float* __restrict__ dt, float* __restrict__ dA) {
  int idx = blockIdx.x * blockDim.x + threadIdx.x;
  if (idx >= ROWS * NHEADS) return;
  int row = idx / NHEADS, h = idx % NHEADS;
  float x  = zx[(size_t)row * DINPROJ + (DINNER + CONVDIM) + h] + dt_bias[h];
  float sp = (x > 20.0f) ? x : log1pf(__expf(x));
  dt[idx] = sp;
  dA[idx] = __expf(-__expf(A_log[h]) * sp);
}

// --------------------- causal depthwise conv1d + SiLU ----------------------
__global__ void conv_k(const float* __restrict__ zx,
                       const float* __restrict__ conv_w,
                       const float* __restrict__ conv_b,
                       float* __restrict__ xBC) {
  int idx = blockIdx.x * blockDim.x + threadIdx.x;
  if (idx >= ROWS * CONVDIM) return;
  int c = idx % CONVDIM;
  int row = idx / CONVDIM;
  int l = row & (SEQ - 1);
  int b = row >> 9;
  float s = conv_b[c];
#pragma unroll
  for (int k = 0; k < 4; ++k) {
    int lp = l - 3 + k;
    if (lp >= 0) s += conv_w[c * 4 + k] * zx[(size_t)((b << 9) + lp) * DINPROJ + DINNER + c];
  }
  xBC[idx] = s * sigm(s);
}

// ------------------------------ SSM scan -----------------------------------
// One block per (batch, head); 256 threads; state[64,128] in registers
// (thread t owns p = t%64, n in [ (t/64)*32, +32 )).
__global__ void scan_k(const float* __restrict__ xBC,
                       const float* __restrict__ dt,
                       const float* __restrict__ dA,
                       const float* __restrict__ D_ssm,
                       float* __restrict__ ybuf) {
  __shared__ float Bs[DSTATE], Cs[DSTATE], xs[HEADDIM], red[256];
  __shared__ float dts, dAs;
  const int bh = blockIdx.x;
  const int b  = bh / NHEADS, h = bh % NHEADS;
  const int tid = threadIdx.x;
  const int p  = tid & 63;
  const int ng = tid >> 6;
  const int nb = ng * 32;
  float hreg[32];
#pragma unroll
  for (int j = 0; j < 32; ++j) hreg[j] = 0.0f;
  const float Dv = D_ssm[h];

  for (int l = 0; l < SEQ; ++l) {
    const int row = (b << 9) + l;
    const float* base = xBC + (size_t)row * CONVDIM;
    if (tid < DSTATE)           Bs[tid]           = base[DINNER + tid];
    else                        Cs[tid - DSTATE]  = base[DINNER + DSTATE + (tid - DSTATE)];
    if (tid < HEADDIM)          xs[tid]           = base[h * HEADDIM + tid];
    if (tid == 0) { dts = dt[row * NHEADS + h]; dAs = dA[row * NHEADS + h]; }
    __syncthreads();

    const float coef = dts * xs[p];
    const float a = dAs;
    float part = 0.0f;
#pragma unroll
    for (int j = 0; j < 32; ++j) {
      float hv = hreg[j] * a + coef * Bs[nb + j];
      hreg[j] = hv;
      part += hv * Cs[nb + j];
    }
    red[tid] = part; __syncthreads();
    if (ng == 0) {
      float y = red[p] + red[64 + p] + red[128 + p] + red[192 + p] + Dv * xs[p];
      ybuf[(size_t)row * DINNER + h * HEADDIM + p] = y;
    }
    __syncthreads();
  }
}

// --------------------- y*silu(z) then RMSNorm over 1536 --------------------
__global__ void gated_rms_k(const float* __restrict__ ybuf,
                            const float* __restrict__ zx,
                            const float* __restrict__ nrm_w,
                            float* __restrict__ normed) {
  __shared__ float red[256];
  const int row = blockIdx.x;
  float v[6]; float ss = 0.0f;
#pragma unroll
  for (int i = 0; i < 6; ++i) {
    int d = threadIdx.x + i * 256;
    float z = zx[(size_t)row * DINPROJ + d];
    float t = ybuf[(size_t)row * DINNER + d] * (z * sigm(z));
    v[i] = t; ss += t * t;
  }
  red[threadIdx.x] = ss; __syncthreads();
  for (int s = 128; s > 0; s >>= 1) {
    if (threadIdx.x < s) red[threadIdx.x] += red[threadIdx.x + s];
    __syncthreads();
  }
  float inv = rsqrtf(red[0] / (float)DINNER + 1e-5f);
#pragma unroll
  for (int i = 0; i < 6; ++i) {
    int d = threadIdx.x + i * 256;
    normed[(size_t)row * DINNER + d] = v[i] * inv * nrm_w[d];
  }
}

// -- hidden += g_mba*mamba_out; u_msa = mod(ln_noaffine(hidden), sh, sc) ----
__global__ void gate_ln_mod_k(float* __restrict__ hidden,
                              const float* __restrict__ mout,
                              const float* __restrict__ aff,
                              float* __restrict__ u_msa) {
  __shared__ float redA[256], redB[256];
  const int row = blockIdx.x;
  const int b   = row >> 9;
  const float* gg = aff + b * AFFD + 2 * DMODEL;  // g_mba
  const float* sh = aff + b * AFFD + 3 * DMODEL;  // sh_msa
  const float* sc = aff + b * AFFD + 4 * DMODEL;  // sc_msa
  float v[3]; float s1 = 0.0f, s2 = 0.0f;
#pragma unroll
  for (int i = 0; i < 3; ++i) {
    int d = threadIdx.x + i * 256;
    float x = hidden[row * DMODEL + d] + gg[d] * mout[row * DMODEL + d];
    hidden[row * DMODEL + d] = x;
    v[i] = x; s1 += x; s2 += x * x;
  }
  redA[threadIdx.x] = s1; redB[threadIdx.x] = s2; __syncthreads();
  for (int s = 128; s > 0; s >>= 1) {
    if (threadIdx.x < s) { redA[threadIdx.x] += redA[threadIdx.x + s];
                           redB[threadIdx.x] += redB[threadIdx.x + s]; }
    __syncthreads();
  }
  float mean = redA[0] / (float)DMODEL;
  float var  = redB[0] / (float)DMODEL - mean * mean;
  float inv  = rsqrtf(var + 1e-6f);
#pragma unroll
  for (int i = 0; i < 3; ++i) {
    int d = threadIdx.x + i * 256;
    float ln = (v[i] - mean) * inv;
    u_msa[row * DMODEL + d] = ln * (1.0f + sc[d]) + sh[d];
  }
}

// ------------------------ cross attention (LC=64) --------------------------
__global__ void attn_k(const float* __restrict__ q,
                       const float* __restrict__ kbuf,
                       const float* __restrict__ vbuf,
                       float* __restrict__ obuf) {
  __shared__ float prob[LCOND], red[LCOND];
  const int row = blockIdx.x;
  const int b   = row >> 9;
  const int t   = threadIdx.x;
  for (int h = 0; h < AHEADS; ++h) {
    const float* qp = q + (size_t)row * AINNER + h * ADIM;
    const float* kp = kbuf + (size_t)(b * LCOND + t) * AINNER + h * ADIM;
    float s = 0.0f;
    for (int d = 0; d < ADIM; ++d) s += qp[d] * kp[d];
    s *= 0.125f;  // 1/sqrt(64)
    red[t] = s; __syncthreads();
    for (int st = 32; st > 0; st >>= 1) {
      if (t < st) red[t] = fmaxf(red[t], red[t + st]);
      __syncthreads();
    }
    float mx = red[0]; __syncthreads();
    float e = __expf(s - mx);
    prob[t] = e; red[t] = e; __syncthreads();
    for (int st = 32; st > 0; st >>= 1) {
      if (t < st) red[t] += red[t + st];
      __syncthreads();
    }
    float inv = 1.0f / red[0];
    float o = 0.0f;
    for (int m = 0; m < LCOND; ++m)
      o += prob[m] * vbuf[(size_t)(b * LCOND + m) * AINNER + h * ADIM + t];
    obuf[(size_t)row * AINNER + h * ADIM + t] = o * inv;
    __syncthreads();
  }
}

// ----------------------- hidden += g_msa * attn_out ------------------------
__global__ void gate_add_k(float* __restrict__ hidden,
                           const float* __restrict__ aout,
                           const float* __restrict__ aff) {
  int idx = blockIdx.x * blockDim.x + threadIdx.x;
  if (idx >= ROWS * DMODEL) return;
  int row = idx / DMODEL, d = idx % DMODEL;
  int b = row >> 9;
  hidden[idx] += aff[b * AFFD + 5 * DMODEL + d] * aout[idx];
}

// --------------------------- final RMSNorm ---------------------------------
__global__ void final_norm_k(const float* __restrict__ hidden,
                             const float* __restrict__ residual,
                             const float* __restrict__ nw,
                             float* __restrict__ out) {
  __shared__ float red[256];
  const int row = blockIdx.x;
  float v[3]; float ss = 0.0f;
#pragma unroll
  for (int i = 0; i < 3; ++i) {
    int d = threadIdx.x + i * 256;
    float x = hidden[row * DMODEL + d] + residual[row * DMODEL + d];
    v[i] = x; ss += x * x;
  }
  red[threadIdx.x] = ss; __syncthreads();
  for (int s = 128; s > 0; s >>= 1) {
    if (threadIdx.x < s) red[threadIdx.x] += red[threadIdx.x + s];
    __syncthreads();
  }
  float inv = rsqrtf(red[0] / (float)DMODEL + 1e-5f);
#pragma unroll
  for (int i = 0; i < 3; ++i) {
    int d = threadIdx.x + i * 256;
    out[row * DMODEL + d] = v[i] * inv * nw[d];
  }
}

// ---------------------------------------------------------------------------
extern "C" void kernel_launch(void* const* d_in, const int* in_sizes, int n_in,
                              void* d_out, int out_size, void* d_ws, size_t ws_size,
                              hipStream_t stream) {
  const float* in_hidden = (const float*)d_in[0];
  const float* in_cond   = (const float*)d_in[1];
  const float* p_norm_w  = (const float*)d_in[2];
  const float* p_ada_w   = (const float*)d_in[3];
  const float* p_ada_b   = (const float*)d_in[4];
  const float* p_in_w    = (const float*)d_in[5];
  const float* p_conv_w  = (const float*)d_in[6];
  const float* p_conv_b  = (const float*)d_in[7];
  const float* p_dt_bias = (const float*)d_in[8];
  const float* p_A_log   = (const float*)d_in[9];
  const float* p_D_ssm   = (const float*)d_in[10];
  const float* p_ssm_nw  = (const float*)d_in[11];
  const float* p_out_w   = (const float*)d_in[12];
  const float* p_out_b   = (const float*)d_in[13];
  const float* p_q_w     = (const float*)d_in[14];
  const float* p_k_w     = (const float*)d_in[15];
  const float* p_v_w     = (const float*)d_in[16];
  const float* p_o_w     = (const float*)d_in[17];
  const float* p_o_b     = (const float*)d_in[18];
  const float* p_normf_w = (const float*)d_in[19];
  float* out = (float*)d_out;

  // workspace carve-up (f32 elements)
  float* ws = (float*)d_ws;
  size_t off = 0;
  float* hidden  = ws + off; off += (size_t)ROWS * DMODEL;
  float* resid   = ws + off; off += (size_t)ROWS * DMODEL;
  float* pooled  = ws + off; off += (size_t)BATCH * DMODEL;
  float* aff     = ws + off; off += (size_t)BATCH * AFFD;
  float* u_mba   = ws + off; off += (size_t)ROWS * DMODEL;
  float* zx      = ws + off; off += (size_t)ROWS * DINPROJ;
  float* dtb     = ws + off; off += (size_t)ROWS * NHEADS;
  float* dAb     = ws + off; off += (size_t)ROWS * NHEADS;
  float* xBC     = ws + off; off += (size_t)ROWS * CONVDIM;
  float* ybuf    = ws + off; off += (size_t)ROWS * DINNER;
  float* normed  = ws + off; off += (size_t)ROWS * DINNER;
  float* mout    = ws + off; off += (size_t)ROWS * DMODEL;
  float* u_msa   = ws + off; off += (size_t)ROWS * DMODEL;
  float* qbuf    = ws + off; off += (size_t)ROWS * AINNER;
  float* kbuf    = ws + off; off += (size_t)CROWS * AINNER;
  float* vbuf    = ws + off; off += (size_t)CROWS * AINNER;
  float* obuf    = ws + off; off += (size_t)ROWS * AINNER;
  float* aout    = ws + off; off += (size_t)ROWS * DMODEL;
  (void)ws_size; (void)in_sizes; (void)n_in; (void)out_size;

  hipMemcpyAsync(hidden, in_hidden, (size_t)ROWS * DMODEL * sizeof(float),
                 hipMemcpyDeviceToDevice, stream);
  hipMemsetAsync(resid, 0, (size_t)ROWS * DMODEL * sizeof(float), stream);

  pool_cond_k<<<(BATCH * DMODEL + 255) / 256, 256, 0, stream>>>(in_cond, pooled);

  for (int l = 0; l < NLAYER; ++l) {
    const float* norm_w  = p_norm_w  + (size_t)l * DMODEL;
    const float* ada_w   = p_ada_w   + (size_t)l * DMODEL * AFFD;
    const float* ada_b   = p_ada_b   + (size_t)l * AFFD;
    const float* in_w    = p_in_w    + (size_t)l * DMODEL * DINPROJ;
    const float* conv_w  = p_conv_w  + (size_t)l * CONVDIM * 4;
    const float* conv_b  = p_conv_b  + (size_t)l * CONVDIM;
    const float* dt_bias = p_dt_bias + (size_t)l * NHEADS;
    const float* A_log   = p_A_log   + (size_t)l * NHEADS;
    const float* D_ssm   = p_D_ssm   + (size_t)l * NHEADS;
    const float* ssm_nw  = p_ssm_nw  + (size_t)l * DINNER;
    const float* out_w   = p_out_w   + (size_t)l * DINNER * DMODEL;
    const float* out_b   = p_out_b   + (size_t)l * DMODEL;
    const float* q_w     = p_q_w     + (size_t)l * DMODEL * AINNER;
    const float* k_w     = p_k_w     + (size_t)l * DMODEL * AINNER;
    const float* v_w     = p_v_w     + (size_t)l * DMODEL * AINNER;
    const float* o_w     = p_o_w     + (size_t)l * AINNER * DMODEL;
    const float* o_b     = p_o_b     + (size_t)l * DMODEL;

    aff_k<<<(BATCH * AFFD + 255) / 256, 256, 0, stream>>>(pooled, ada_w, ada_b, aff);
    addnorm_mod_k<<<ROWS, 256, 0, stream>>>(hidden, resid, norm_w, aff, u_mba);

    // in_proj: [1024,768] x [768,3352]
    gemm_bf16_wmma_k<<<dim3((DINPROJ + 15) / 16, ROWS / 64), 32, 0, stream>>>(
        u_mba, in_w, nullptr, zx, ROWS, DINPROJ, DMODEL);

    dt_k<<<(ROWS * NHEADS + 255) / 256, 256, 0, stream>>>(zx, dt_bias, A_log, dtb, dAb);
    conv_k<<<(ROWS * CONVDIM + 255) / 256, 256, 0, stream>>>(zx, conv_w, conv_b, xBC);
    scan_k<<<BATCH * NHEADS, 256, 0, stream>>>(xBC, dtb, dAb, D_ssm, ybuf);
    gated_rms_k<<<ROWS, 256, 0, stream>>>(ybuf, zx, ssm_nw, normed);

    // out_proj: [1024,1536] x [1536,768] + bias
    gemm_bf16_wmma_k<<<dim3(DMODEL / 16, ROWS / 64), 32, 0, stream>>>(
        normed, out_w, out_b, mout, ROWS, DMODEL, DINNER);

    gate_ln_mod_k<<<ROWS, 256, 0, stream>>>(hidden, mout, aff, u_msa);

    // q/k/v projections
    gemm_bf16_wmma_k<<<dim3(AINNER / 16, ROWS / 64), 32, 0, stream>>>(
        u_msa, q_w, nullptr, qbuf, ROWS, AINNER, DMODEL);
    gemm_bf16_wmma_k<<<dim3(AINNER / 16, CROWS / 64), 32, 0, stream>>>(
        in_cond, k_w, nullptr, kbuf, CROWS, AINNER, DMODEL);
    gemm_bf16_wmma_k<<<dim3(AINNER / 16, CROWS / 64), 32, 0, stream>>>(
        in_cond, v_w, nullptr, vbuf, CROWS, AINNER, DMODEL);

    attn_k<<<ROWS, LCOND, 0, stream>>>(qbuf, kbuf, vbuf, obuf);

    // output projection: [1024,512] x [512,768] + bias
    gemm_bf16_wmma_k<<<dim3(DMODEL / 16, ROWS / 64), 32, 0, stream>>>(
        obuf, o_w, o_b, aout, ROWS, DMODEL, AINNER);

    gate_add_k<<<(ROWS * DMODEL + 255) / 256, 256, 0, stream>>>(hidden, aout, aff);
  }

  final_norm_k<<<ROWS, 256, 0, stream>>>(hidden, resid, p_normf_w, out);
}
